// DeltaProductLayer_4286377361863
// MI455X (gfx1250) — matrix operand order repaired and checked by reading, not compiled, not run on hardware
//
#include <hip/hip_runtime.h>
#include <hip/hip_bf16.h>
#include <stdint.h>

// ---------------- problem constants ----------------
#define B_ 2
#define S_ 1024
#define F_ 1024
#define H_ 8
#define D_ 64
#define R_ 2
#define M_   (B_ * S_)            // 2048 GEMM rows
#define NQ_  (H_ * D_)            // 512
#define NK_  (H_ * R_ * D_)       // 1024
#define NB_  (H_ * R_)            // 16
#define NCAT (NQ_ + 2 * NK_ + NB_) // 2576
#define NP_  2624                 // padded to 41*64 so GEMM needs no edge predication
#define QOFF 0
#define KOFF NQ_                  // 512
#define VOFF (NQ_ + NK_)          // 1536
#define BOFF (NQ_ + 2 * NK_)     // 2560

// ---------------- WMMA fragment types (gfx1250, wave32) ----------------
typedef __attribute__((ext_vector_type(16))) __bf16 bf16x16;
typedef __attribute__((ext_vector_type(8)))  float  f32x8;

union FragAB { bf16x16 v; uint4 u4[2]; };
union FragC  { f32x8   v; float f[8];  };

__device__ __forceinline__ unsigned short f2bf(float f) {
  unsigned int u = __float_as_uint(f);
  unsigned int r = u + 0x7FFFu + ((u >> 16) & 1u);   // round-to-nearest-even
  return (unsigned short)(r >> 16);
}

__device__ __forceinline__ float fast_sigmoid(float x) {
  return __builtin_amdgcn_rcpf(1.f + __expf(-x));    // v_rcp_f32, no div expansion
}

// gfx1250 async memory->LDS copy (ASYNCcnt path), b128 per lane.
__device__ __forceinline__ void async_load_b128(unsigned lds_off, const void* gaddr) {
  asm volatile("global_load_async_to_lds_b128 %0, %1, off"
               :: "v"(lds_off), "v"(gaddr) : "memory");
}
__device__ __forceinline__ void wait_asynccnt0() {
  asm volatile("s_wait_asynccnt 0x0" ::: "memory");
}
__device__ __forceinline__ void wait_asynccnt3() {   // keep the 3 just-issued in flight
  asm volatile("s_wait_asynccnt 0x3" ::: "memory");
}

// ---------------- pack kernels (f32 -> bf16) ----------------
__global__ void pack_bf16_kernel(const float* __restrict__ in,
                                 unsigned short* __restrict__ out, int n) {
  int i = blockIdx.x * 256 + threadIdx.x;
  if (i < n) out[i] = f2bf(in[i]);
}

// Transposed+concatenated weights: out[n*F_ + f] = bf16(Wcat[f][n]), n in [0,NP_).
// Column-major weights make the GEMM B-operand K-contiguous per column, so the
// B tile is async-copied straight into its final LDS layout (no scatter).
__global__ void pack_wcat_t_kernel(const float* __restrict__ Wq,
                                   const float* __restrict__ Wk,
                                   const float* __restrict__ Wv,
                                   const float* __restrict__ Wb,
                                   unsigned short* __restrict__ out) {
  int i = blockIdx.x * 256 + threadIdx.x;   // i = n*F_ + f (writes coalesced)
  if (i >= NP_ * F_) return;
  int n = i / F_, f = i % F_;
  float v = 0.f;
  if      (n < KOFF) v = Wq[f * NQ_ + n];
  else if (n < VOFF) v = Wk[f * NK_ + (n - KOFF)];
  else if (n < BOFF) v = Wv[f * NK_ + (n - VOFF)];
  else if (n < NCAT) v = Wb[f * NB_ + (n - BOFF)];
  out[i] = f2bf(v);
}

// Generic transpose pack: out[n*K + k] = bf16(in[k*N + n]).
__global__ void pack_t_kernel(const float* __restrict__ in,
                              unsigned short* __restrict__ out, int K, int N) {
  int i = blockIdx.x * 256 + threadIdx.x;   // i = n*K + k
  if (i >= K * N) return;
  int n = i / K, k = i % K;
  out[i] = f2bf(in[(size_t)k * N + n]);
}

// ---------------- bf16 WMMA GEMM, double-buffered async-LDS pipeline ----------
// C[M,N] = act(A[M,K] @ Bt[N,K]^T + bias).  cols < silu_end -> silu,
// cols >= sig_start -> sigmoid, else identity. M mult 128, N mult 64, K mult 32.
#define BM 128
#define BN 64
#define BK 32
#define LDA 40   // ushorts per A row in LDS (32 + 8 pad -> conflict-free b128 reads)
#define LDB 40   // ushorts per B column in LDS

__device__ __forceinline__ float gemm_epilogue(float x, int col, int silu_end,
                                               int sig_start,
                                               const float* __restrict__ bias) {
  if (bias) x += bias[col];
  if (col < silu_end)        x = x * fast_sigmoid(x);  // silu
  else if (col >= sig_start) x = fast_sigmoid(x);      // sigmoid
  return x;
}

__global__ __launch_bounds__(256)
void gemm_bf16_kernel(const unsigned short* __restrict__ A,   // [M,K] bf16
                      const unsigned short* __restrict__ Bt,  // [N,K] bf16 (B^T)
                      float* __restrict__ C,                  // [M,N] f32
                      const float* __restrict__ bias,         // [N] or nullptr
                      int M, int N, int K, int silu_end, int sig_start) {
  __shared__ __align__(16) unsigned short sA[2][BM * LDA];
  __shared__ __align__(16) unsigned short sB[2][BN * LDB];

  const int bm   = blockIdx.y * BM;
  const int bn   = blockIdx.x * BN;
  const int tid  = threadIdx.x;
  const int lane = tid & 31;
  const int wave = tid >> 5;
  const int wr   = wave & 3;   // 4 row tiles of 32
  const int wc   = wave >> 2;  // 2 col tiles of 32

  FragC acc00, acc01, acc10, acc11;
#pragma unroll
  for (int i = 0; i < 8; i++) {
    acc00.f[i] = 0.f; acc01.f[i] = 0.f; acc10.f[i] = 0.f; acc11.f[i] = 0.f;
  }

  // A tile 128x32 = 4096 bf16: thread copies 2 x b128 (row tid>>1, chunk (tid&1)*16, +8)
  const int arow = tid >> 1, achk = (tid & 1) * 16;
  // B tile 64 cols x 32 K = 2048 bf16: thread copies 1 x b128 (col tid>>2, K (tid&3)*8)
  const int bcol = tid >> 2, bchk = (tid & 3) * 8;

  const int half16 = lane >> 4;
  const int lrow   = lane & 15;

  const unsigned ldsA[2] = {
    (unsigned)(uintptr_t)(&sA[0][arow * LDA + achk]),
    (unsigned)(uintptr_t)(&sA[1][arow * LDA + achk])};
  const unsigned ldsB[2] = {
    (unsigned)(uintptr_t)(&sB[0][bcol * LDB + bchk]),
    (unsigned)(uintptr_t)(&sB[1][bcol * LDB + bchk])};
  const unsigned short* gA = A  + (size_t)(bm + arow) * K + achk;
  const unsigned short* gB = Bt + (size_t)(bn + bcol) * K + bchk;

  const int KT = K / BK;
  // prologue: tile 0 in flight
  async_load_b128(ldsA[0],      gA);
  async_load_b128(ldsA[0] + 16, gA + 8);
  async_load_b128(ldsB[0],      gB);

  for (int kt = 0; kt < KT; ++kt) {
    const int cur = kt & 1, nxt = cur ^ 1;
    const bool more = (kt + 1) < KT;
    if (more) {  // overlap next tile's memory->LDS with this tile's WMMAs
      int k1 = (kt + 1) * BK;
      async_load_b128(ldsA[nxt],      gA + k1);
      async_load_b128(ldsA[nxt] + 16, gA + k1 + 8);
      async_load_b128(ldsB[nxt],      gB + k1);
      if (kt + 2 < KT) __builtin_prefetch(gA + (kt + 2) * BK, 0, 0);
      wait_asynccnt3();   // tile kt landed; 3 newer copies still in flight
    } else {
      wait_asynccnt0();
    }
    __syncthreads();      // tile kt visible to all waves

    const unsigned short* a = sA[cur];
    const unsigned short* b = sB[cur];
    // A fragments 16x32: lanes 0-15 row=l, K={0..7,16..23}; lanes 16-31 K={8..15,24..31}
    FragAB af0, af1;
    {
      int r0 = wr * 32 + lrow;
      int kb = half16 * 8;
      af0.u4[0] = *(const uint4*)(a + r0 * LDA + kb);
      af0.u4[1] = *(const uint4*)(a + r0 * LDA + kb + 16);
      int r1 = r0 + 16;
      af1.u4[0] = *(const uint4*)(a + r1 * LDA + kb);
      af1.u4[1] = *(const uint4*)(a + r1 * LDA + kb + 16);
    }
    // B fragments 32x16: lanes 0-15 K={0..7,16..23}, lanes 16-31 K={8..15,24..31}, col = lane%16
    FragAB bf0, bf1;
    {
      int kb = half16 * 8;
      int c0 = wc * 32 + lrow;
      bf0.u4[0] = *(const uint4*)(b + c0 * LDB + kb);
      bf0.u4[1] = *(const uint4*)(b + c0 * LDB + kb + 16);
      int c1 = c0 + 16;
      bf1.u4[0] = *(const uint4*)(b + c1 * LDB + kb);
      bf1.u4[1] = *(const uint4*)(b + c1 * LDB + kb + 16);
    }
    acc00.v = __builtin_amdgcn_wmma_f32_16x16x32_bf16(
        false, af0.v, false, bf0.v, (short)0, acc00.v, false, false);
    acc01.v = __builtin_amdgcn_wmma_f32_16x16x32_bf16(
        false, af0.v, false, bf1.v, (short)0, acc01.v, false, false);
    acc10.v = __builtin_amdgcn_wmma_f32_16x16x32_bf16(
        false, af1.v, false, bf0.v, (short)0, acc10.v, false, false);
    acc11.v = __builtin_amdgcn_wmma_f32_16x16x32_bf16(
        false, af1.v, false, bf1.v, (short)0, acc11.v, false, false);
    __syncthreads();      // all reads of buf `cur` done before it is rewritten
  }

  // D layout: elem r -> row = r + 8*(lane>=16), col = lane%16
  const int r0base = bm + wr * 32 + half16 * 8;
  const int r1base = r0base + 16;
  const int cc0    = bn + wc * 32 + lrow;
  const int cc1    = cc0 + 16;
#pragma unroll
  for (int r = 0; r < 8; r++) {
    int row0 = r0base + r, row1 = r1base + r;
    C[(size_t)row0 * N + cc0] = gemm_epilogue(acc00.f[r], cc0, silu_end, sig_start, bias);
    C[(size_t)row0 * N + cc1] = gemm_epilogue(acc01.f[r], cc1, silu_end, sig_start, bias);
    C[(size_t)row1 * N + cc0] = gemm_epilogue(acc10.f[r], cc0, silu_end, sig_start, bias);
    C[(size_t)row1 * N + cc1] = gemm_epilogue(acc11.f[r], cc1, silu_end, sig_start, bias);
  }
}

// ---------------- L2-normalize q and k rows (in place, post-silu) --------------
__global__ __launch_bounds__(32)
void normalize_kernel(float* __restrict__ qkvb) {
  int vid  = blockIdx.x;
  int lane = threadIdx.x;
  float* p;
  if (vid < M_ * H_) {
    int m = vid / H_, h = vid % H_;
    p = qkvb + (size_t)m * NP_ + QOFF + h * D_;
  } else {
    int t = vid - M_ * H_;
    int m = t / (H_ * R_), hr = t % (H_ * R_);
    p = qkvb + (size_t)m * NP_ + KOFF + hr * D_;
  }
  float e0 = p[lane], e1 = p[lane + 32];
  float ss = e0 * e0 + e1 * e1;
#pragma unroll
  for (int mset = 16; mset > 0; mset >>= 1) ss += __shfl_xor(ss, mset, 32);
  float inv = __builtin_amdgcn_rcpf(sqrtf(ss) + 1e-6f);
  p[lane]      = e0 * inv;
  p[lane + 32] = e1 * inv;
}

// ---------------- sequential delta-product scan -------------------------------
// T_s = keep_s * T_{s-1} @ A_s + B_s realized as two fused delta-rule rank-1
// updates: T <- keep*T; T += b1*(v1 - T k1) k1^T; T += b2*(v2 - T k2) k2^T.
// One wave per (b,h); lane owns state rows `lane` and `lane+32` in VGPRs.
__global__ __launch_bounds__(32)
void scan_kernel(const float* __restrict__ qkvb,
                 const unsigned char* __restrict__ mask,   // [B_*S_] bool
                 const float* __restrict__ carry,          // [B_,H_,D_,D_]
                 float* __restrict__ x,                    // [M_, NQ_]
                 float* __restrict__ carry_out) {          // [B_,H_,D_,D_]
  const int bh = blockIdx.x, b = bh / H_, h = bh % H_;
  const int lane = threadIdx.x;
  __shared__ __align__(16) float sq[64], sk1[64], sk2[64], sv1[64], sv2[64];

  float T0[64], T1[64];
  const float* c0 = carry + (size_t)bh * D_ * D_;
#pragma unroll
  for (int j = 0; j < 64; j++) {
    T0[j] = c0[lane * 64 + j];
    T1[j] = c0[(lane + 32) * 64 + j];
  }
  const int qo  = QOFF + h * D_;
  const int k1o = KOFF + (h * R_ + 0) * D_;
  const int k2o = KOFF + (h * R_ + 1) * D_;
  const int v1o = VOFF + (h * R_ + 0) * D_;
  const int v2o = VOFF + (h * R_ + 1) * D_;
  const int bo  = BOFF + h * R_;

  for (int s = 0; s < S_; s++) {
    size_t m = (size_t)b * S_ + s;
    const float* row = qkvb + m * NP_;
    sq [lane] = row[qo  + lane];  sq [lane + 32] = row[qo  + lane + 32];
    sk1[lane] = row[k1o + lane];  sk1[lane + 32] = row[k1o + lane + 32];
    sk2[lane] = row[k2o + lane];  sk2[lane + 32] = row[k2o + lane + 32];
    sv1[lane] = row[v1o + lane];  sv1[lane + 32] = row[v1o + lane + 32];
    sv2[lane] = row[v2o + lane];  sv2[lane + 32] = row[v2o + lane + 32];
    float b1 = row[bo + 0], b2 = row[bo + 1];      // post-sigmoid
    float keep = 1.0f - (float)mask[m];
    __syncthreads();

    float u0 = 0.f, u1 = 0.f;
#pragma unroll
    for (int j = 0; j < 64; j++) {
      T0[j] *= keep; T1[j] *= keep;
      u0 = fmaf(T0[j], sk1[j], u0); u1 = fmaf(T1[j], sk1[j], u1);
    }
    float w0 = b1 * (sv1[lane] - u0), w1 = b1 * (sv1[lane + 32] - u1);
    float p0 = 0.f, p1 = 0.f;
#pragma unroll
    for (int j = 0; j < 64; j++) {
      T0[j] = fmaf(w0, sk1[j], T0[j]); T1[j] = fmaf(w1, sk1[j], T1[j]);
      p0 = fmaf(T0[j], sk2[j], p0);    p1 = fmaf(T1[j], sk2[j], p1);
    }
    float y0 = b2 * (sv2[lane] - p0), y1 = b2 * (sv2[lane + 32] - p1);
    float x0 = 0.f, x1 = 0.f;
#pragma unroll
    for (int j = 0; j < 64; j++) {
      T0[j] = fmaf(y0, sk2[j], T0[j]); T1[j] = fmaf(y1, sk2[j], T1[j]);
      x0 = fmaf(T0[j], sq[j], x0);     x1 = fmaf(T1[j], sq[j], x1);
    }
    x[m * NQ_ + h * D_ + lane]      = x0;
    x[m * NQ_ + h * D_ + lane + 32] = x1;
    __syncthreads();
  }
  float* co = carry_out + (size_t)bh * D_ * D_;
#pragma unroll
  for (int j = 0; j < 64; j++) {
    co[lane * 64 + j]        = T0[j];
    co[(lane + 32) * 64 + j] = T1[j];
  }
}

// ---------------- RMS norm + scale, emit bf16 for final GEMM ------------------
__global__ __launch_bounds__(32)
void rms_kernel(const float* __restrict__ x, const float* __restrict__ rms_scale,
                unsigned short* __restrict__ xn) {
  int m = blockIdx.x, lane = threadIdx.x;
  float vals[16];
  float ss = 0.f;
#pragma unroll
  for (int t = 0; t < 16; t++) {
    float v = x[(size_t)m * NQ_ + lane + t * 32];
    vals[t] = v;
    ss = fmaf(v, v, ss);
  }
#pragma unroll
  for (int mset = 16; mset > 0; mset >>= 1) ss += __shfl_xor(ss, mset, 32);
  float r = rsqrtf(ss * (1.f / (float)NQ_) + 1e-6f);
#pragma unroll
  for (int t = 0; t < 16; t++) {
    int c = lane + t * 32;
    xn[(size_t)m * NQ_ + c] = f2bf(vals[t] * r * rms_scale[c]);
  }
}

// ---------------- host-side launch --------------------------------------------
extern "C" void kernel_launch(void* const* d_in, const int* in_sizes, int n_in,
                              void* d_out, int out_size, void* d_ws, size_t ws_size,
                              hipStream_t stream) {
  const float*         inputs   = (const float*)d_in[0];
  const unsigned char* mask     = (const unsigned char*)d_in[1]; // bool array
  const float*         carry    = (const float*)d_in[2];
  const float*         Wq       = (const float*)d_in[3];
  const float*         Wk       = (const float*)d_in[4];
  const float*         Wv       = (const float*)d_in[5];
  const float*         Wb       = (const float*)d_in[6];
  const float*         rmsScale = (const float*)d_in[7];
  const float*         Wo       = (const float*)d_in[8];
  const float*         bo       = (const float*)d_in[9];

  // workspace layout (all 16B aligned)
  char* ws = (char*)d_ws;
  size_t off = 0;
  unsigned short* inb   = (unsigned short*)(ws + off); off += (size_t)M_ * F_ * 2;     // 4 MB
  unsigned short* wcatT = (unsigned short*)(ws + off); off += (size_t)NP_ * F_ * 2;    // 5.4 MB
  unsigned short* wobT  = (unsigned short*)(ws + off); off += (size_t)F_ * NQ_ * 2;    // 1 MB
  float*          qkvb  = (float*)(ws + off);          off += (size_t)M_ * NP_ * 4;    // 21.5 MB
  float*          xbuf  = (float*)(ws + off);          off += (size_t)M_ * NQ_ * 4;    // 4 MB
  unsigned short* xn    = (unsigned short*)(ws + off); off += (size_t)M_ * NQ_ * 2;    // 2 MB

  float* carry_out = (float*)d_out;                  // B_*H_*D_*D_ = 65536 floats
  float* y         = (float*)d_out + (size_t)B_ * H_ * D_ * D_;  // [M_, F_]

  // 1) pack bf16 operands (weights transposed for async B-tile staging)
  pack_bf16_kernel<<<(M_ * F_ + 255) / 256, 256, 0, stream>>>(inputs, inb, M_ * F_);
  pack_wcat_t_kernel<<<(NP_ * F_ + 255) / 256, 256, 0, stream>>>(Wq, Wk, Wv, Wb, wcatT);
  pack_t_kernel<<<(F_ * NQ_ + 255) / 256, 256, 0, stream>>>(Wo, wobT, NQ_, F_);

  // 2) fused projection GEMM: qkvb = act(inputs @ [Wq|Wk|Wv|Wb])
  {
    dim3 grid(NP_ / BN, M_ / BM);  // 41 x 16
    gemm_bf16_kernel<<<grid, 256, 0, stream>>>(inb, wcatT, qkvb, nullptr,
                                               M_, NP_, F_, VOFF, BOFF);
  }

  // 3) L2-normalize q and k heads in place
  normalize_kernel<<<M_ * (H_ + H_ * R_), 32, 0, stream>>>(qkvb);

  // 4) sequential delta-product scan (one wave per (b,h))
  scan_kernel<<<B_ * H_, 32, 0, stream>>>(qkvb, mask, carry, xbuf, carry_out);

  // 5) RMS norm -> bf16
  rms_kernel<<<M_, 32, 0, stream>>>(xbuf, rmsScale, xn);

  // 6) output GEMM: y = xn @ Wo + bo
  {
    dim3 grid(F_ / BN, M_ / BM);   // 16 x 16
    gemm_bf16_kernel<<<grid, 256, 0, stream>>>(xn, wobT, y, bo,
                                               M_, F_, NQ_, 0, 1 << 30);
  }
}
